// CropRoi_19739669693063
// MI455X (gfx1250) — compile-verified, matrix operand order checked by reading, not run on hardware
//
#include <hip/hip_runtime.h>

// ---------------------------------------------------------------------------
// CropRoi fused pipeline for MI455X (gfx1250, wave32, WMMA, 320KB LDS/WGP)
//
// One workgroup (256 threads = 8 wave32) per proposal (64 proposals).
//
//  * conv1x1 commutes with trilinear upsample -> GEMM1 runs on the 8^3 crop.
//  * IN1/IN2 folded to per-channel scale/shift; IN2 stats from a first WMMA
//    pass (deterministic staged reduction); second WMMA pass normalizes and
//    max-pools via LDS u32 atomicMax on non-negative float bits.
//  * All WMMA operands stored in LDS pre-swizzled into fragment order:
//    tile(16x32 f16) = 512 contiguous f16, per-lane 16 contiguous f16 (32B)
//    -> fragment load is one 32B vector LDS load instead of 16x ds_load_u16.
// ---------------------------------------------------------------------------

typedef _Float16 v16h __attribute__((ext_vector_type(16)));
typedef float    v8f  __attribute__((ext_vector_type(8)));

#define NPROP 64
#define CIN   128
#define CMID  64
#define S4V   512      // 8^3
#define CHUNK 512      // spatial chunk for GEMM2 (2 z-slices of 16x16)
#define NCHUNK 8
#define POOLV 343      // 7^3

struct __align__(32) SMem {
  _Float16 cat[CIN * CHUNK];    // 128KB: swizzled GEMM B operand (fe1, then cat chunks)
  _Float16 up[CMID * S4V];      // 64KB : up_small = conv1(fe1) on 8^3 grid (linear [m][pos])
  _Float16 w2[CMID * CIN];      // 16KB : w_back2, swizzled (GEMM2 A)
  union {                       // 87.8KB, phase-disjoint:
    _Float16 wA[CMID * CIN];    //   GEMM1 A (w_up2), swizzled
    float    stage[2 * 2048];   //   pass-A per-lane stat partials (sum, sumsq)
    unsigned pool[CMID * POOLV];//   pass-B pooled result (float bits, >=0)
  } u;
  int   ilo[16], ihi[16];       // trilinear taps (align_corners, 8->16)
  float iw[16];
  int   binlo[16], bincnt[16];  // adaptive-pool 16->7 bin membership
  float b1[CMID], b2[CMID];     // conv biases
  float sc1[CMID], sh1[CMID];   // IN1 folded scale/shift
  float sc2[CMID], sh2[CMID];   // IN2 folded scale/shift
  float part1[512];             // IN1 stat partials (256 sum + 256 sumsq)
};

__device__ __forceinline__ v8f wmma_step(v16h a, v16h b, v8f c) {
  // D = A(16x32 f16) * B(32x16 f16) + C(16x16 f32)
  return __builtin_amdgcn_wmma_f32_16x16x32_f16(
      /*neg_a=*/false, a, /*neg_b=*/false, b,
      /*c_mod=*/(short)0, c, /*reuse_a=*/false, /*reuse_b=*/false);
}

// CDNA5 16-bit operand fragment layout (one 16x32 tile = 512 f16):
//   lane L holds row (L&15), k-half (L>>4); element i -> kl = (i&7)+16*(i>>3)+8*(L>>4)
// Inverse: (rowInTile, kl) -> flat index within tile (lane-major, 16 f16/lane).
__device__ __forceinline__ int frag_idx(int rowInTile, int kl) {
  const int L = rowInTile + (((kl >> 3) & 1) << 4);
  const int e = (kl & 7) + ((kl >> 4) << 3);
  return L * 16 + e;
}
// A operand (weights [M=64][K=128]): tile grid 4x4, tile (mt,kt) at (mt*4+kt)*512
__device__ __forceinline__ int swzA(int m, int k) {
  return ((m >> 4) * 4 + (k >> 5)) * 512 + frag_idx(m & 15, k & 31);
}
// B operand ([K=128][N=512]): tile grid 4x32, tile (kt,nt) at (kt*32+nt)*512
__device__ __forceinline__ int swzB(int k, int n) {
  return ((k >> 5) * 32 + (n >> 4)) * 512 + frag_idx(n & 15, k & 31);
}

__device__ __forceinline__ v16h ld_frag(const _Float16* base) {
  return *(const v16h*)base;   // 32B contiguous, 32B aligned -> vector LDS load
}

// Trilinear (align_corners) sample of channel k of up_small at 16^3 pos.
__device__ __forceinline__ float tri_corners(const _Float16* u,
                                             int o000, int o001, int o010, int o011,
                                             int o100, int o101, int o110, int o111,
                                             float wx, float wy, float wz) {
  const float c000 = (float)u[o000], c001 = (float)u[o001];
  const float c010 = (float)u[o010], c011 = (float)u[o011];
  const float c100 = (float)u[o100], c101 = (float)u[o101];
  const float c110 = (float)u[o110], c111 = (float)u[o111];
  const float c00 = c000 + wx * (c001 - c000);
  const float c01 = c010 + wx * (c011 - c010);
  const float c10 = c100 + wx * (c101 - c100);
  const float c11 = c110 + wx * (c111 - c110);
  const float c0 = c00 + wy * (c01 - c00);
  const float c1 = c10 + wy * (c11 - c10);
  return c0 + wz * (c1 - c0);
}

__global__ __launch_bounds__(256, 1)
void crop_roi_kernel(const float* __restrict__ out1,   // [2][64][64][64][64]
                     const float* __restrict__ comb2,  // [2][128][32][32][32]
                     const int*   __restrict__ props,  // [64][7] (i32)
                     const float* __restrict__ w_up2, const float* __restrict__ b_up2,
                     const float* __restrict__ g_up2, const float* __restrict__ be_up2,
                     const float* __restrict__ w_back2, const float* __restrict__ b_back2,
                     const float* __restrict__ g_back2, const float* __restrict__ be_back2,
                     float* __restrict__ out)          // [64][64][7][7][7]
{
  __shared__ SMem sm;
  const int tid  = threadIdx.x;
  const int lane = tid & 31;
  const int wave = tid >> 5;
  const int p    = blockIdx.x;

  const int pb = props[p * 7 + 0];
  const int z0 = props[p * 7 + 1];
  const int y0 = props[p * 7 + 2];
  const int x0 = props[p * 7 + 3];
  const int z4 = z0 >> 2, y4 = y0 >> 2, x4 = x0 >> 2;
  const int z2 = z0 >> 1, y2 = y0 >> 1, x2 = x0 >> 1;

  // -------- phase 0: tables, biases, weights (swizzled), fe1 crop ----------
  if (tid < 16) {
    const float posf = ((float)tid * 7.0f) / 15.0f;   // align_corners map 16->8
    int lo = (int)posf;
    if (lo > 7) lo = 7;
    sm.ilo[tid] = lo;
    sm.ihi[tid] = (lo + 1 < 7) ? lo + 1 : 7;
    sm.iw[tid]  = posf - (float)lo;
    int b0 = 0, cnt = 0;                               // adaptive pool 16->7 bins
    for (int i = 0; i < 7; ++i) {
      const int s = (i * 16) / 7;
      const int e = ((i + 1) * 16 + 6) / 7;
      if (s <= tid && tid < e) { if (!cnt) b0 = i; ++cnt; }
    }
    sm.binlo[tid] = b0;
    sm.bincnt[tid] = cnt;
  }
  if (tid < CMID) { sm.b1[tid] = b_up2[tid]; sm.b2[tid] = b_back2[tid]; }
  for (int i = tid; i < CMID * CIN; i += 256) {
    const int m = i >> 7, k = i & 127;
    const int d = swzA(m, k);
    sm.u.wA[d] = (_Float16)w_up2[i];
    sm.w2[d]   = (_Float16)w_back2[i];
  }
  {
    const float* cb = comb2 + (size_t)pb * CIN * 32768;
    for (int i = tid; i < CIN * S4V; i += 256) {       // c=K, l=N (z*64+y*8+x)
      const int c = i >> 9, l = i & 511;
      const int z = l >> 6, y = (l >> 3) & 7, x = l & 7;
      sm.cat[swzB(c, l)] =
          (_Float16)cb[(size_t)c * 32768 +
                       (size_t)(z4 + z) * 1024 + (y4 + y) * 32 + (x4 + x)];
    }
  }
  __syncthreads();

  // -------- phase 1: GEMM1  up_small[64,512] = w_up2[64,128] @ fe1[128,512] --
  const int mtile = wave & 3;
  const int m0 = mtile << 4;
  {
    v16h a1[4];
#pragma unroll
    for (int kk = 0; kk < 4; ++kk)
      a1[kk] = ld_frag(&sm.u.wA[(mtile * 4 + kk) * 512 + lane * 16]);
    for (int j = 0; j < 16; ++j) {
      const int nt = (wave >> 2) + 2 * j;
      v8f acc = {};
#pragma unroll
      for (int kk = 0; kk < 4; ++kk)
        acc = wmma_step(a1[kk], ld_frag(&sm.cat[(kk * 32 + nt) * 512 + lane * 16]), acc);
      union { v8f v; float e[8]; } d; d.v = acc;
      const int n  = (nt << 4) + (lane & 15);
      const int mh = m0 + ((lane >> 4) << 3);
#pragma unroll
      for (int r = 0; r < 8; ++r) {
        const int m = mh + r;
        sm.up[m * S4V + n] = (_Float16)(d.e[r] + sm.b1[m]);
      }
    }
  }
  __syncthreads();

  // -------- phase 2: IN1 stats over the (virtual) upsampled 16^3 field ------
  {
    const int k = tid >> 2, q = tid & 3;                // 4 threads / channel
    const _Float16* u = sm.up + (k << 9);
    float s = 0.f, sq = 0.f;
    for (int pos = q * 1024; pos < (q + 1) * 1024; ++pos) {
      const int z = pos >> 8, y = (pos >> 4) & 15, x = pos & 15;
      const int zl = sm.ilo[z] * 64, yl = sm.ilo[y] * 8, xl = sm.ilo[x];
      const int zh = sm.ihi[z] * 64, yh = sm.ihi[y] * 8, xh = sm.ihi[x];
      const float v = tri_corners(u, zl+yl+xl, zl+yl+xh, zl+yh+xl, zl+yh+xh,
                                     zh+yl+xl, zh+yl+xh, zh+yh+xl, zh+yh+xh,
                                  sm.iw[x], sm.iw[y], sm.iw[z]);
      s += v; sq += v * v;
    }
    sm.part1[tid] = s;
    sm.part1[256 + tid] = sq;
  }
  __syncthreads();
  if (tid < CMID) {
    float S = 0.f, Q = 0.f;
    for (int q = 0; q < 4; ++q) { S += sm.part1[tid * 4 + q]; Q += sm.part1[256 + tid * 4 + q]; }
    const float mean = S * (1.0f / 4096.0f);
    const float var  = Q * (1.0f / 4096.0f) - mean * mean;
    const float sc   = rsqrtf(var + 1e-5f) * g_up2[tid];
    sm.sc1[tid] = sc;
    sm.sh1[tid] = be_up2[tid] - mean * sc;
  }
  __syncthreads();

  const float* ob = out1 + (size_t)pb * CMID * 262144;

  // Chunk build: rows 0..63 = relu(IN1(upsample)), rows 64..127 = o1 crop.
  // Positions: l = zloc*256 + y*16 + x, global z = cn*2 + zloc.
  auto build_chunk = [&](int cn) {
    // upsampled rows: each thread owns 2 spatial positions across all 64 ch
    for (int half = 0; half < 2; ++half) {
      const int l   = tid + half * 256;
      const int pos = cn * CHUNK + l;
      const int z = pos >> 8, y = (pos >> 4) & 15, x = pos & 15;
      const int zl = sm.ilo[z] * 64, yl = sm.ilo[y] * 8, xl = sm.ilo[x];
      const int zh = sm.ihi[z] * 64, yh = sm.ihi[y] * 8, xh = sm.ihi[x];
      const float wx = sm.iw[x], wy = sm.iw[y], wz = sm.iw[z];
      const int o000 = zl+yl+xl, o001 = zl+yl+xh, o010 = zl+yh+xl, o011 = zl+yh+xh;
      const int o100 = zh+yl+xl, o101 = zh+yl+xh, o110 = zh+yh+xl, o111 = zh+yh+xh;
      for (int k = 0; k < CMID; ++k) {
        float v = tri_corners(sm.up + (k << 9), o000,o001,o010,o011,
                              o100,o101,o110,o111, wx, wy, wz);
        v = fmaxf(v * sm.sc1[k] + sm.sh1[k], 0.f);
        sm.cat[swzB(k, l)] = (_Float16)v;
      }
    }
    // o1 rows: 2048 x-runs of 16 floats (one 64B line each), float2 vectorized
    for (int rr = tid; rr < CMID * 2 * 16; rr += 256) {
      const int c = rr >> 5, zloc = (rr >> 4) & 1, y = rr & 15;
      const float* g = ob + (size_t)c * 262144 +
                       (size_t)(z2 + cn * 2 + zloc) * 4096 + (y2 + y) * 64 + x2;
      if (cn + 1 < NCHUNK) __builtin_prefetch(g + 2 * 4096, 0, 0);  // next chunk line
      const int lb = zloc * 256 + y * 16;
#pragma unroll
      for (int xx = 0; xx < 8; ++xx) {
        const float2 v = *(const float2*)(g + 2 * xx);   // x2 even -> 8B aligned
        sm.cat[swzB(64 + c, lb + 2 * xx)]     = (_Float16)v.x;
        sm.cat[swzB(64 + c, lb + 2 * xx + 1)] = (_Float16)v.y;
      }
    }
  };

  // -------- phase 3: GEMM2 pass A (gather IN2 stats, deterministic) ---------
  float lsum[8], lsq[8];
#pragma unroll
  for (int r = 0; r < 8; ++r) { lsum[r] = 0.f; lsq[r] = 0.f; }
  {
    v16h a2[4];
#pragma unroll
    for (int kk = 0; kk < 4; ++kk)
      a2[kk] = ld_frag(&sm.w2[(mtile * 4 + kk) * 512 + lane * 16]);
    for (int cn = 0; cn < NCHUNK; ++cn) {
      build_chunk(cn);
      __syncthreads();
      for (int j = 0; j < 16; ++j) {
        const int nt = (wave >> 2) + 2 * j;
        v8f acc = {};
#pragma unroll
        for (int kk = 0; kk < 4; ++kk)
          acc = wmma_step(a2[kk], ld_frag(&sm.cat[(kk * 32 + nt) * 512 + lane * 16]), acc);
        union { v8f v; float e[8]; } d; d.v = acc;
        const int mh = m0 + ((lane >> 4) << 3);
#pragma unroll
        for (int r = 0; r < 8; ++r) {
          const float v = d.e[r] + sm.b2[mh + r];
          lsum[r] += v; lsq[r] += v * v;
        }
      }
      __syncthreads();
    }
  }
  {
    const int base = (wave * 32 + lane) * 8;
#pragma unroll
    for (int r = 0; r < 8; ++r) {
      sm.u.stage[base + r] = lsum[r];
      sm.u.stage[2048 + base + r] = lsq[r];
    }
  }
  __syncthreads();
  if (tid < CMID) {   // channel m: contributions from 2 waves x 16 lanes, fixed order
    const int mt = tid >> 4, half = (tid >> 3) & 1, r = tid & 7;
    float S = 0.f, Q = 0.f;
    for (int w = mt; w < 8; w += 4)
      for (int l = 0; l < 16; ++l) {
        const int idx = (w * 32 + (half * 16 + l)) * 8 + r;
        S += sm.u.stage[idx];
        Q += sm.u.stage[2048 + idx];
      }
    const float mean = S * (1.0f / 4096.0f);
    const float var  = Q * (1.0f / 4096.0f) - mean * mean;
    const float sc   = rsqrtf(var + 1e-5f) * g_back2[tid];
    sm.sc2[tid] = sc;
    sm.sh2[tid] = be_back2[tid] - mean * sc;
  }
  __syncthreads();
  for (int i = tid; i < CMID * POOLV; i += 256) sm.u.pool[i] = 0u;  // relu >= 0
  __syncthreads();

  // -------- phase 4: GEMM2 pass B (normalize + relu + max-pool scatter) -----
  {
    v16h a2[4];
#pragma unroll
    for (int kk = 0; kk < 4; ++kk)
      a2[kk] = ld_frag(&sm.w2[(mtile * 4 + kk) * 512 + lane * 16]);
    for (int cn = 0; cn < NCHUNK; ++cn) {
      build_chunk(cn);
      __syncthreads();
      for (int j = 0; j < 16; ++j) {
        const int nt = (wave >> 2) + 2 * j;
        v8f acc = {};
#pragma unroll
        for (int kk = 0; kk < 4; ++kk)
          acc = wmma_step(a2[kk], ld_frag(&sm.cat[(kk * 32 + nt) * 512 + lane * 16]), acc);
        union { v8f v; float e[8]; } d; d.v = acc;
        const int mh  = m0 + ((lane >> 4) << 3);
        const int pos = cn * CHUNK + (nt << 4) + (lane & 15);
        const int z = pos >> 8, y = (pos >> 4) & 15, x = pos & 15;
        const int bz = sm.binlo[z], by = sm.binlo[y], bx = sm.binlo[x];
        const int cz = sm.bincnt[z], cy = sm.bincnt[y], cx = sm.bincnt[x];
#pragma unroll
        for (int r = 0; r < 8; ++r) {
          const int m = mh + r;
          const float v = fmaxf((d.e[r] + sm.b2[m]) * sm.sc2[m] + sm.sh2[m], 0.f);
          const unsigned bits = __float_as_uint(v); // v>=0: uint order == float order
          unsigned* pc = &sm.u.pool[m * POOLV];
          for (int iz = 0; iz < cz; ++iz)
            for (int iy = 0; iy < cy; ++iy)
              for (int ix = 0; ix < cx; ++ix)
                atomicMax(&pc[(bz + iz) * 49 + (by + iy) * 7 + (bx + ix)], bits);
        }
      }
      __syncthreads();
    }
  }

  // -------- phase 5: writeback [64,7,7,7] -----------------------------------
  for (int i = tid; i < CMID * POOLV; i += 256)
    out[(size_t)p * (CMID * POOLV) + i] = __uint_as_float(sm.u.pool[i]);
}

extern "C" void kernel_launch(void* const* d_in, const int* in_sizes, int n_in,
                              void* d_out, int out_size, void* d_ws, size_t ws_size,
                              hipStream_t stream) {
  (void)in_sizes; (void)n_in; (void)out_size; (void)d_ws; (void)ws_size;
  const float* out1    = (const float*)d_in[2];
  const float* comb2   = (const float*)d_in[3];
  const int*   props   = (const int*)d_in[4];
  const float* w_up2   = (const float*)d_in[5];
  const float* b_up2   = (const float*)d_in[6];
  const float* g_up2   = (const float*)d_in[7];
  const float* be_up2  = (const float*)d_in[8];
  const float* w_back2 = (const float*)d_in[9];
  const float* b_back2 = (const float*)d_in[10];
  const float* g_back2 = (const float*)d_in[11];
  const float* be_back2= (const float*)d_in[12];
  float* out = (float*)d_out;

  crop_roi_kernel<<<NPROP, 256, 0, stream>>>(
      out1, comb2, props, w_up2, b_up2, g_up2, be_up2,
      w_back2, b_back2, g_back2, be_back2, out);
}
